// ThreeDRegisterNetwork_82265803587893
// MI455X (gfx1250) — matrix operand order repaired and checked by reading, not compiled, not run on hardware
//
#include <hip/hip_runtime.h>
#include <hip/hip_bf16.h>
#include <math.h>
#include <stdint.h>

typedef __attribute__((ext_vector_type(16))) _Float16 v16h;
typedef __attribute__((ext_vector_type(8)))  float    v8f;

static constexpr int cH = 296, cW = 296, cHW = cH * cW;
static constexpr int cPH = cH + 2, cPW = cW + 2;            // zero-halo padded activation layout
static constexpr int cB = 2, cE = 64, cNV = 2500, cNP = 5023, cM = 16384, cK = 11;

// ---------------- Stage A: projection / gather / top-K / scatter ----------------

__global__ void k_project(const float* __restrict__ tp, const float* __restrict__ tf,
                          const int* __restrict__ vis, const float* __restrict__ embed,
                          const float* __restrict__ filler,
                          int* __restrict__ pix, float* __restrict__ feats) {
  int gid = blockIdx.x * blockDim.x + threadIdx.x;
  if (gid >= cB * cNV) return;
  int b = gid / cNV;
  int vp = vis[gid];
  float x = 0.f, y = 0.f, z = 0.f;
  if (vp < cNP) { const float* p = tp + ((size_t)b * cNP + vp) * 3; x = p[0]; y = p[1]; z = p[2]; }
  const float* T = tf + b * 16;
  float cx = T[0]*x + T[1]*y + T[2]*z  + T[3];
  float cy = T[4]*x + T[5]*y + T[6]*z  + T[7];
  float cz = T[8]*x + T[9]*y + T[10]*z + T[11];
  float inv = 1.0f / (cz + 1e-8f);
  float sx = (cx * inv * 12.0f + 1.0f) * (cW * 0.5f);
  float sy = (cy * inv * 12.0f + 1.0f) * (cH * 0.5f);
  int px = (int)rintf(sx), py = (int)rintf(sy);
  bool invalid = (px < 0) | (px > cW - 1) | (py < 0) | (py > cH - 1);
  if (invalid) { px = 0; py = 0; }
  pix[gid * 2 + 0] = px; pix[gid * 2 + 1] = py;
  float* fo = feats + (size_t)gid * cE;
  for (int e = 0; e < cE; ++e) {
    float fe = (vp >= cNP) ? 0.0f : embed[(size_t)vp * cE + e];
    fo[e] = invalid ? filler[e] : fe;
  }
}

__global__ void k_fill(float* __restrict__ reg, const float* __restrict__ filler) {
  size_t i = (size_t)blockIdx.x * blockDim.x + threadIdx.x;
  if (i < (size_t)cB * cHW * cE) reg[i] = filler[i & (cE - 1)];
}

__global__ void k_topk_scatter(const int* __restrict__ apm, const int* __restrict__ pix,
                               const float* __restrict__ feats, float* __restrict__ reg) {
  int gid = blockIdx.x * blockDim.x + threadIdx.x;
  if (gid >= cB * cM) return;
  int b = gid / cM;
  int ax = apm[gid * 2 + 0], ay = apm[gid * 2 + 1];
  if (ax < 0 || ax > cW - 1 || ay < 0 || ay > cH - 1) { ax = 0; ay = 0; }
  float dk[cK]; int ik[cK];
  for (int k = 0; k < cK; ++k) { dk[k] = 3.0e38f; ik[k] = 0; }
  bool inproj = false;
  const int* pp = pix + (size_t)b * cNV * 2;
  for (int s = 0; s < cNV; ++s) {
    int px = pp[s * 2], py = pp[s * 2 + 1];
    if (px == ax && py == ay) inproj = true;
    float dx = (float)(ax - px), dy = (float)(ay - py);
    float d = sqrtf(dx * dx + dy * dy);
    if (d < dk[cK - 1]) {
      int j = cK - 1;
      while (j > 0 && dk[j - 1] > d) { dk[j] = dk[j - 1]; ik[j] = ik[j - 1]; --j; }
      dk[j] = d; ik[j] = s;
    }
  }
  float* dst = reg + ((size_t)b * cHW + (size_t)ay * cW + ax) * cE;
  if (inproj) { for (int e = 0; e < cE; ++e) dst[e] = 0.0f; return; }
  float wk[cK], wsum = 0.f;
  for (int k = 0; k < cK; ++k) { wk[k] = 1.0f / (dk[k] + 1e-8f); wsum += wk[k]; }
  float winv = 1.0f / wsum;
  for (int e = 0; e < cE; ++e) {
    float a = 0.f;
    for (int k = 0; k < cK; ++k) a += feats[((size_t)b * cNV + ik[k]) * cE + e] * wk[k];
    dst[e] = a * winv;
  }
}

__global__ void k_scatter_feats(const int* __restrict__ pix, const float* __restrict__ feats,
                                float* __restrict__ reg) {
  int gid = blockIdx.x * blockDim.x + threadIdx.x;
  if (gid >= cB * cNV) return;
  int b = gid / cNV;
  int px = pix[gid * 2], py = pix[gid * 2 + 1];
  float* dst = reg + ((size_t)b * cHW + (size_t)py * cW + px) * cE;
  const float* src = feats + (size_t)gid * cE;
  for (int e = 0; e < cE; ++e) dst[e] = src[e];
}

__global__ void k_set00(float* __restrict__ reg, const float* __restrict__ filler) {
  reg[(size_t)blockIdx.x * cHW * cE + threadIdx.x] = filler[threadIdx.x];
}

// reg (unpadded NHWC f32) -> padded-halo NHWC f16
__global__ void k_cvt(const float* __restrict__ reg, _Float16* __restrict__ act, size_t n) {
  size_t i = (size_t)blockIdx.x * blockDim.x + threadIdx.x;
  if (i >= n) return;
  int e = (int)(i & (cE - 1));
  size_t p = i >> 6;                // pixel id
  int b = (int)(p / cHW);
  int r = (int)(p - (size_t)b * cHW);
  int y = r / cW, x = r - y * cW;
  act[(((size_t)b * cPH + (y + 1)) * cPW + (x + 1)) * cE + e] = (_Float16)reg[i];
}

// zero the 1-pixel halo of a padded activation tensor with channel count C
__global__ void k_zero_border(_Float16* __restrict__ act, int C) {
  const int perB = 2 * cPW + 2 * (cPH - 2);
  int gid = blockIdx.x * blockDim.x + threadIdx.x;
  if (gid >= cB * perB * C) return;
  int e = gid % C; int t = gid / C;
  int b = t / perB; int r = t - b * perB;
  int y, x;
  if (r < cPW)            { y = 0;        x = r; }
  else if (r < 2 * cPW)   { y = cPH - 1;  x = r - cPW; }
  else { int rr = r - 2 * cPW; y = 1 + (rr >> 1); x = (rr & 1) ? (cPW - 1) : 0; }
  act[(((size_t)b * cPH + y) * cPW + x) * C + e] = (_Float16)0.f;
}

// ---------------- weight pack + BN fold ----------------

// OIHW f32 -> [Co][(ky*KS+kx)*Cin + ci] f16 (GEMM-B rows = out channel, cols = K)
__global__ void k_wpack(const float* __restrict__ w, _Float16* __restrict__ wp,
                        int Cin, int Cout, int KS) {
  int idx = blockIdx.x * blockDim.x + threadIdx.x;
  int Ktot = KS * KS * Cin;
  if (idx >= Cout * Ktot) return;
  int co = idx / Ktot, k = idx - co * Ktot;
  int ci = k % Cin; int q = k / Cin; int kx = q % KS; int ky = q / KS;
  wp[idx] = (_Float16)w[(((size_t)co * Cin + ci) * KS + ky) * KS + kx];
}

__global__ void k_fold(const float* __restrict__ bias, const float* __restrict__ g,
                       const float* __restrict__ be, const float* __restrict__ mean,
                       const float* __restrict__ var, float* __restrict__ s,
                       float* __restrict__ t, int C, int bn) {
  int c = blockIdx.x * blockDim.x + threadIdx.x;
  if (c >= C) return;
  if (bn) { float sc = g[c] * rsqrtf(var[c] + 1e-5f); s[c] = sc; t[c] = (bias[c] - mean[c]) * sc + be[c]; }
  else    { s[c] = 1.0f; t[c] = bias[c]; }
}

// ---------------- implicit-GEMM conv: async-LDS double-buffered WMMA pipeline ----------------
// Block = 256 threads = 8 wave32. Tile: 128 px x 64 outch; each wave 16 px x 64 ch.
// Per K-step each thread issues exactly 3 GLOBAL_LOAD_ASYNC_TO_LDS_B128 (wave-uniform
// ASYNCcnt), computes the other buffer, then s_wait_asynccnt 0 + one barrier.
__global__ __launch_bounds__(256) void k_conv_wmma(
    const _Float16* __restrict__ act, const _Float16* __restrict__ wp,
    const float* __restrict__ sc, const float* __restrict__ bi,
    _Float16* __restrict__ outh, float* __restrict__ outA,
    float* __restrict__ outB, const float* __restrict__ featAdd,
    int Cin, int Cout, int KS, int relu) {
  __shared__ __align__(16) _Float16 sAt[2 * 128 * 40];  // 2 x (128 px rows x 32 K halves, stride 40)
  __shared__ __align__(16) _Float16 sBt[2 * 64 * 40];   // 2 x (64 outch rows x 32 K halves)

  const int b = blockIdx.z;
  const int p0 = blockIdx.x * 128;
  const int co0 = blockIdx.y * 64;
  const int tid = threadIdx.x;
  const int wv = tid >> 5, lane = tid & 31;
  const int m0 = wv * 16;
  const int pad = (KS == 3) ? 1 : 0;
  const int padoff = 1 - pad;               // halo layout correction
  const int Ktot = KS * KS * Cin;
  const int nK = Ktot >> 5;

  // A staging: thread -> (row, 16-half chunk). Pixel clamped so halo reads stay in-buffer.
  const int ar = tid >> 1;
  const int ah = (tid & 1) << 4;
  const int ap0 = p0 + ar;
  const int ap = (ap0 < cHW) ? ap0 : (cHW - 1);
  const int ayy = ap / cW, axx = ap - ayy * cW;
  // B staging: thread -> (outch row, 8-half chunk)
  const int brow = tid >> 2;
  const int bk = (tid & 3) << 3;

  const uint64_t actB64 = (uint64_t)(uintptr_t)act;
  const uint64_t wpB64  = (uint64_t)(uintptr_t)wp;

  auto stage = [&](int ksx, int bufsel) {
    int kk = ksx << 5;
    int q = kk / Cin;
    int ci = kk - q * Cin;
    int kx = q % KS, ky = q / KS;
    int iy = ayy + ky + padoff, ix = axx + kx + padoff;
    unsigned aoff = (unsigned)(((((b * cPH + iy) * cPW) + ix) * Cin + ci + ah) * 2);
    unsigned ldsA = (unsigned)(uintptr_t)(sAt + bufsel * (128 * 40) + ar * 40 + ah);
    asm volatile("global_load_async_to_lds_b128 %0, %1, %2\n\t"
                 "global_load_async_to_lds_b128 %0, %1, %2 offset:32"
                 :: "v"(ldsA), "v"(aoff), "s"(actB64) : "memory");
    unsigned boff = (unsigned)((((co0 + brow) * Ktot) + kk + bk) * 2);
    unsigned ldsB = (unsigned)(uintptr_t)(sBt + bufsel * (64 * 40) + brow * 40 + bk);
    asm volatile("global_load_async_to_lds_b128 %0, %1, %2"
                 :: "v"(ldsB), "v"(boff), "s"(wpB64) : "memory");
  };

  v8f acc0 = {}, acc1 = {}, acc2 = {}, acc3 = {};

  const int kbA = (lane & 16);        // 0 / 16 (contiguous k = 16h + s, same order as B)
  const int kbB = (lane & 16);        // 0 / 16 (contiguous k = 16h + s, same order as A)
  const int arow = m0 + (lane & 15);
  const int ncol = lane & 15;

  stage(0, 0);
  asm volatile("s_wait_asynccnt 0x0" ::: "memory");
  __syncthreads();

  for (int ks = 0; ks < nK; ++ks) {
    const int cur = ks & 1;
    if (ks + 1 < nK) stage(ks + 1, cur ^ 1);   // prefetch next tile into other buffer

    union Frag { uint4 u[2]; v16h h; };
    Frag fa, f0, f1, f2, f3;
    const _Float16* sa = sAt + cur * (128 * 40);
    const _Float16* sb = sBt + cur * (64 * 40);
    const uint4* pa = (const uint4*)(sa + arow * 40 + kbA);
    fa.u[0] = pa[0]; fa.u[1] = pa[1];
    const uint4* pb0 = (const uint4*)(sb + (0 * 16 + ncol) * 40 + kbB);
    const uint4* pb1 = (const uint4*)(sb + (1 * 16 + ncol) * 40 + kbB);
    const uint4* pb2 = (const uint4*)(sb + (2 * 16 + ncol) * 40 + kbB);
    const uint4* pb3 = (const uint4*)(sb + (3 * 16 + ncol) * 40 + kbB);
    f0.u[0] = pb0[0]; f0.u[1] = pb0[1];
    f1.u[0] = pb1[0]; f1.u[1] = pb1[1];
    f2.u[0] = pb2[0]; f2.u[1] = pb2[1];
    f3.u[0] = pb3[0]; f3.u[1] = pb3[1];

    acc0 = __builtin_amdgcn_wmma_f32_16x16x32_f16(false, fa.h, false, f0.h, (short)0, acc0, false, false);
    acc1 = __builtin_amdgcn_wmma_f32_16x16x32_f16(false, fa.h, false, f1.h, (short)0, acc1, false, false);
    acc2 = __builtin_amdgcn_wmma_f32_16x16x32_f16(false, fa.h, false, f2.h, (short)0, acc2, false, false);
    acc3 = __builtin_amdgcn_wmma_f32_16x16x32_f16(false, fa.h, false, f3.h, (short)0, acc3, false, false);

    asm volatile("s_wait_asynccnt 0x0" ::: "memory");  // next tile landed in LDS
    __syncthreads();                                    // publish to all waves
  }

  // ---- fused epilogue: BN/bias (+ReLU), optional f32 NCHW + residual + f16 NHWC(halo) ----
  const int mrow = (lane & 16) >> 1;  // C/D: VGPR r -> M = r + 8*(lane>=16)
  v8f accs[4] = {acc0, acc1, acc2, acc3};
#pragma unroll
  for (int nb = 0; nb < 4; ++nb) {
    int co = co0 + nb * 16 + ncol;
    float s = sc[co], t = bi[co];
#pragma unroll
    for (int r = 0; r < 8; ++r) {
      int p = p0 + m0 + mrow + r;
      if (p >= cHW) continue;
      float v = accs[nb][r] * s + t;
      if (relu) v = fmaxf(v, 0.0f);
      int py = p / cW, px = p - py * cW;
      size_t nchw = (((size_t)b * Cout + co) * cH + py) * cW + px;
      if (outA) outA[nchw] = v;
      float hv = v;
      if (featAdd) hv += featAdd[nchw];
      if (outB) outB[nchw] = hv;
      if (outh) outh[(((size_t)b * cPH + (py + 1)) * cPW + (px + 1)) * Cout + co] = (_Float16)hv;
    }
  }
}

// ---------------- reduction for sum(out) ----------------

__global__ void k_zero(float* p) { if (threadIdx.x == 0 && blockIdx.x == 0) *p = 0.0f; }

__global__ void k_sum(const float* __restrict__ x, float* __restrict__ out, size_t n) {
  __shared__ float sh[256];
  size_t i = (size_t)blockIdx.x * blockDim.x + threadIdx.x;
  size_t stride = (size_t)gridDim.x * blockDim.x;
  float a = 0.f;
  for (; i < n; i += stride) a += x[i];
  sh[threadIdx.x] = a;
  __syncthreads();
  for (int s = 128; s > 0; s >>= 1) {
    if (threadIdx.x < s) sh[threadIdx.x] += sh[threadIdx.x + s];
    __syncthreads();
  }
  if (threadIdx.x == 0) atomicAdd(out, sh[0]);
}

// ---------------- host orchestration ----------------

extern "C" void kernel_launch(void* const* d_in, const int* in_sizes, int n_in,
                              void* d_out, int out_size, void* d_ws, size_t ws_size,
                              hipStream_t stream) {
  (void)in_sizes; (void)n_in; (void)out_size; (void)ws_size;
  // input flattening order (setup_inputs dict, params recursed in insertion order):
  // 0 features, 1 target_points, 2 transform, 3 embed, 4 filler,
  // 5..28 convert[L]{w,b,g,be,m,v}, 29..36 outproc[L]{w,b}, 37 visible_points, 38 all_pts_mask
  const float* features  = (const float*)d_in[0];
  const float* tpoints   = (const float*)d_in[1];
  const float* transform = (const float*)d_in[2];
  const float* embed     = (const float*)d_in[3];
  const float* filler    = (const float*)d_in[4];
  const int*   vis       = (const int*)d_in[37];
  const int*   apm       = (const int*)d_in[38];
  auto conv_in = [&](int L, int part) { return (const float*)d_in[5 + L * 6 + part]; };
  auto outp_in = [&](int L, int part) { return (const float*)d_in[29 + L * 2 + part]; };

  char* ws = (char*)d_ws;
  size_t off = 0;
  auto walloc = [&](size_t bytes) { size_t o = off; off = (off + bytes + 255) & ~(size_t)255; return o; };
  int*      pix   = (int*)(ws + walloc((size_t)cB * cNV * 2 * sizeof(int)));
  float*    feats = (float*)(ws + walloc((size_t)cB * cNV * cE * sizeof(float)));
  float*    reg   = (float*)(ws + walloc((size_t)cB * cHW * cE * sizeof(float)));
  _Float16* actA  = (_Float16*)(ws + walloc((size_t)cB * cPH * cPW * 512 * sizeof(_Float16)));
  _Float16* actB  = (_Float16*)(ws + walloc((size_t)cB * cPH * cPW * 512 * sizeof(_Float16)));
  _Float16* wpack = (_Float16*)(ws + walloc((size_t)512 * 512 * 9 * sizeof(_Float16)));
  float*    scl   = (float*)(ws + walloc(512 * sizeof(float)));
  float*    bia   = (float*)(ws + walloc(512 * sizeof(float)));

  size_t Nout = (size_t)cB * 256 * cHW;
  float* d_proc = (float*)d_out;          // output 0: proc
  float* d_outT = d_proc + Nout;          // output 1: out
  float* d_outF = d_outT + Nout;          // output 2: out + features
  float* d_sum  = d_outF + Nout;          // output 3: sum(out)

  const int borderN = cB * (2 * cPW + 2 * (cPH - 2));

  // ---- Stage A ----
  k_project<<<(cB * cNV + 255) / 256, 256, 0, stream>>>(tpoints, transform, vis, embed, filler, pix, feats);
  {
    size_t n = (size_t)cB * cHW * cE;
    k_fill<<<(unsigned)((n + 255) / 256), 256, 0, stream>>>(reg, filler);
  }
  k_topk_scatter<<<(cB * cM + 255) / 256, 256, 0, stream>>>(apm, pix, feats, reg);
  k_scatter_feats<<<(cB * cNV + 255) / 256, 256, 0, stream>>>(pix, feats, reg);
  k_set00<<<cB, cE, 0, stream>>>(reg, filler);
  {
    size_t n = (size_t)cB * cHW * cE;
    k_cvt<<<(unsigned)((n + 255) / 256), 256, 0, stream>>>(reg, actA, n);
  }

  // ---- convert_net: 4x conv3x3 + BN + ReLU ----
  const int cins[4]  = {64, 512, 512, 256};
  const int couts[4] = {512, 512, 256, 256};
  _Float16* src = actA; _Float16* dst = actB;
  for (int L = 0; L < 4; ++L) {
    int Cin = cins[L], Cout = couts[L], Ktot = 9 * Cin;
    k_wpack<<<(Cout * Ktot + 255) / 256, 256, 0, stream>>>(conv_in(L, 0), wpack, Cin, Cout, 3);
    k_fold<<<(Cout + 255) / 256, 256, 0, stream>>>(conv_in(L, 1), conv_in(L, 2), conv_in(L, 3),
                                                   conv_in(L, 4), conv_in(L, 5), scl, bia, Cout, 1);
    k_zero_border<<<(borderN * Cin + 255) / 256, 256, 0, stream>>>(src, Cin);
    dim3 grid((cHW + 127) / 128, Cout / 64, cB);
    if (L < 3)
      k_conv_wmma<<<grid, 256, 0, stream>>>(src, wpack, scl, bia, dst, nullptr, nullptr, nullptr,
                                            Cin, Cout, 3, 1);
    else  // last layer: write `out` (f32 NCHW), `out+features` (f32 NCHW + f16 halo-NHWC)
      k_conv_wmma<<<grid, 256, 0, stream>>>(src, wpack, scl, bia, dst, d_outT, d_outF, features,
                                            Cin, Cout, 3, 1);
    _Float16* t = src; src = dst; dst = t;
  }

  // ---- sum(out) ----
  k_zero<<<1, 1, 0, stream>>>(d_sum);
  k_sum<<<1024, 256, 0, stream>>>(d_outT, d_sum, Nout);

  // ---- outproc: 3x conv3x3(+ReLU) + 1x conv1x1 ----  (src = out+features, f16 halo-NHWC)
  for (int L = 0; L < 3; ++L) {
    int Ktot = 9 * 256;
    k_wpack<<<(256 * Ktot + 255) / 256, 256, 0, stream>>>(outp_in(L, 0), wpack, 256, 256, 3);
    k_fold<<<1, 256, 0, stream>>>(outp_in(L, 1), outp_in(L, 1), outp_in(L, 1),
                                  outp_in(L, 1), outp_in(L, 1), scl, bia, 256, 0);
    k_zero_border<<<(borderN * 256 + 255) / 256, 256, 0, stream>>>(src, 256);
    dim3 grid((cHW + 127) / 128, 256 / 64, cB);
    k_conv_wmma<<<grid, 256, 0, stream>>>(src, wpack, scl, bia, dst, nullptr, nullptr, nullptr,
                                          256, 256, 3, 1);
    _Float16* t = src; src = dst; dst = t;
  }
  {
    int Ktot = 256;  // 1x1 (reads interior only; halo irrelevant but zeroed path unused)
    k_wpack<<<(256 * Ktot + 255) / 256, 256, 0, stream>>>(outp_in(3, 0), wpack, 256, 256, 1);
    k_fold<<<1, 256, 0, stream>>>(outp_in(3, 1), outp_in(3, 1), outp_in(3, 1),
                                  outp_in(3, 1), outp_in(3, 1), scl, bia, 256, 0);
    dim3 grid((cHW + 127) / 128, 256 / 64, cB);
    k_conv_wmma<<<grid, 256, 0, stream>>>(src, wpack, scl, bia, nullptr, d_proc, nullptr, nullptr,
                                          256, 256, 1, 0);
  }
}
